// MultiNetwork_37873021616617
// MI455X (gfx1250) — matrix operand order repaired and checked by reading, not compiled, not run on hardware
//
#include <hip/hip_runtime.h>
#include <hip/hip_bf16.h>
#include <stdint.h>

// ---------------------------------------------------------------------------
// Ensemble MLP (32 nets, 256->512->512->128, batch 4096) fused WMMA kernel
// for gfx1250 (CDNA5, wave32).
//   * bf16 -> v_wmma_f32_16x16x32_bf16 with fp32 accumulation
//   * h1/h2 intermediates live in LDS (never touch HBM)
//   * weight panels staged global->LDS ONCE per block (vs 8x per-wave reads)
//     via async global_load_async_to_lds_b128, double-buffered (ASYNCcnt)
// ---------------------------------------------------------------------------

typedef __attribute__((ext_vector_type(16))) __bf16 v16bf;
typedef __attribute__((ext_vector_type(8)))  float  v8f;

struct I4x2 { int4 a, b; };   // 32 bytes -> bit_cast to v16bf

__device__ __forceinline__ unsigned int f2bf(float f) {
  unsigned int u = __builtin_bit_cast(unsigned int, f);
  u += 0x7FFFu + ((u >> 16) & 1u);   // round-to-nearest-even
  return u >> 16;
}

// -------------------- fp32 -> bf16 weight conversion -----------------------
__global__ void cvt_f32_to_bf16(const float* __restrict__ in,
                                uint2* __restrict__ out, int n4) {
  int i = blockIdx.x * blockDim.x + threadIdx.x;
  if (i >= n4) return;
  float4 f = ((const float4*)in)[i];
  uint2 o;
  o.x = f2bf(f.x) | (f2bf(f.y) << 16);
  o.y = f2bf(f.z) | (f2bf(f.w) << 16);
  out[i] = o;
}

// --------------------------- fused MLP kernel ------------------------------
#define BT        128                 // batch rows per block
#define X_STRIDE  528                 // bytes/row, x tile   (256*2 + 16 pad)
#define H_STRIDE  1040                // bytes/row, h1/h2    (512*2 + 16 pad)
#define STAGE_OFF (2 * BT * H_STRIDE) // 266,240
#define STAGE_STRIDE 1040             // bytes/row, weight panel (16B pad)
#define STAGE_BUF (16 * STAGE_STRIDE) // 16,640 per buffer
#define LDS_BYTES (STAGE_OFF + 2 * STAGE_BUF)  // 299,520 <= 320KB WGP LDS

// Issue async copies of one 16-row weight panel (rows [rowBase,rowBase+16) of
// a [Nout x K] bf16 matrix) into an LDS stage buffer. 256 threads, coalesced.
// cshift: log2(16B-chunks per row) = log2(K/8).
__device__ __forceinline__ void stage_issue(const unsigned short* __restrict__ W,
                                            int rowBase, int K, int cshift,
                                            char* buf, int tid) {
  const int iters = (16 << cshift) >> 8;          // 2 for K=256, 4 for K=512
  const unsigned lds0 = (unsigned)(uintptr_t)buf; // LDS byte offset = addr[31:0]
  for (int it = 0; it < iters; ++it) {
    int q = it * 256 + tid;
    int r = q >> cshift;
    int c = q & ((1 << cshift) - 1);
    const void* g = (const void*)(W + (size_t)(rowBase + r) * K + c * 8);
    unsigned lds = lds0 + r * STAGE_STRIDE + c * 16;
    asm volatile("global_load_async_to_lds_b128 %0, %1, off"
                 :: "v"(lds), "v"(g) : "memory");
  }
}

__device__ __forceinline__ void wait_async0() {
  asm volatile("s_wait_asynccnt 0x0" ::: "memory");
}

// B fragment (32x16 bf16, K x N) from a staged LDS panel row pointer.
__device__ __forceinline__ v16bf bfrag_lds(const char* bufRow, int kk) {
  I4x2 p;
  p.a = *(const int4*)(bufRow + kk * 64);
  p.b = *(const int4*)(bufRow + kk * 64 + 32);
  return __builtin_bit_cast(v16bf, p);
}

__global__ __launch_bounds__(256, 1)
void mlp_ens_kernel(const float* __restrict__ x,
                    const unsigned short* __restrict__ W1b,
                    const float* __restrict__ b1,
                    const unsigned short* __restrict__ W2b,
                    const float* __restrict__ b2,
                    const unsigned short* __restrict__ W3b,
                    const float* __restrict__ b3,
                    float* __restrict__ out) {
  extern __shared__ char smem[];
  char* r0 = smem;                    // x tile, later h2 tile
  char* r1 = smem + BT * H_STRIDE;    // h1 tile
  char* st = smem + STAGE_OFF;        // 2 weight-panel stage buffers

  const int tid    = threadIdx.x;
  const int wv     = tid >> 5;        // wave id (0..7), wave32
  const int lane   = tid & 31;
  const int lanelo = lane & 15;
  const int laneh  = lane >> 4;
  const int net    = blockIdx.y;
  const int rb     = blockIdx.x * BT;
  const int wrow   = wv * 16;         // this wave's row stripe in the tile

  const unsigned short* Wn1 = W1b + (size_t)net * 512 * 256;
  const unsigned short* Wn2 = W2b + (size_t)net * 512 * 512;
  const unsigned short* Wn3 = W3b + (size_t)net * 128 * 512;

  // ---- Phase 0: cooperative load of x tile [128x256] fp32 -> bf16 LDS ----
  for (int it = 0; it < 32; ++it) {
    int idx = it * 256 + tid;         // float4 index, 64 per row
    int r   = idx >> 6;
    int c4  = idx & 63;
    float4 f = ((const float4*)(x + (size_t)(rb + r) * 256))[c4];
    uint2 o;
    o.x = f2bf(f.x) | (f2bf(f.y) << 16);
    o.y = f2bf(f.z) | (f2bf(f.w) << 16);
    *(uint2*)(r0 + r * X_STRIDE + c4 * 8) = o;
  }
  __syncthreads();

  // ===================== Layer 1: K=256, Nout=512 =====================
  stage_issue(Wn1, 0, 256, 5, st, tid);            // panel 0 -> buf 0
  v16bf a1[8];
  {
    const char* base = r0 + (wrow + lanelo) * X_STRIDE + laneh * 16;
#pragma unroll
    for (int kk = 0; kk < 8; ++kk) {
      I4x2 p;
      p.a = *(const int4*)(base + kk * 64);
      p.b = *(const int4*)(base + kk * 64 + 32);
      a1[kk] = __builtin_bit_cast(v16bf, p);
    }
  }
  wait_async0();
  __syncthreads();
  for (int ct = 0; ct < 32; ++ct) {
    if (ct + 1 < 32)
      stage_issue(Wn1, (ct + 1) * 16, 256, 5, st + ((ct + 1) & 1) * STAGE_BUF, tid);
    const int nb = ct * 16;
    const char* bufRow =
        st + (ct & 1) * STAGE_BUF + lanelo * STAGE_STRIDE + laneh * 16;
    v8f acc = {};
#pragma unroll
    for (int kk = 0; kk < 8; ++kk)
      acc = __builtin_amdgcn_wmma_f32_16x16x32_bf16(
          false, a1[kk], false, bfrag_lds(bufRow, kk), (short)0, acc, false, false);
    const float bias = b1[net * 512 + nb + lanelo];
#pragma unroll
    for (int v = 0; v < 8; ++v) {
      float val = fmaxf(acc[v] + bias, 0.0f);
      *(unsigned short*)(r1 + (wrow + v + 8 * laneh) * H_STRIDE +
                         (nb + lanelo) * 2) = (unsigned short)f2bf(val);
    }
    wait_async0();
    __syncthreads();   // panel swap; also fences x-region reads vs h2 writes
  }

  // ===================== Layer 2: K=512, Nout=512 =====================
  stage_issue(Wn2, 0, 512, 6, st, tid);
  v16bf a2[16];
  {
    const char* base = r1 + (wrow + lanelo) * H_STRIDE + laneh * 16;
#pragma unroll
    for (int kk = 0; kk < 16; ++kk) {
      I4x2 p;
      p.a = *(const int4*)(base + kk * 64);
      p.b = *(const int4*)(base + kk * 64 + 32);
      a2[kk] = __builtin_bit_cast(v16bf, p);
    }
  }
  wait_async0();
  __syncthreads();
  for (int ct = 0; ct < 32; ++ct) {
    if (ct + 1 < 32)
      stage_issue(Wn2, (ct + 1) * 16, 512, 6, st + ((ct + 1) & 1) * STAGE_BUF, tid);
    const int nb = ct * 16;
    const char* bufRow =
        st + (ct & 1) * STAGE_BUF + lanelo * STAGE_STRIDE + laneh * 16;
    v8f acc = {};
#pragma unroll
    for (int kk = 0; kk < 16; ++kk)
      acc = __builtin_amdgcn_wmma_f32_16x16x32_bf16(
          false, a2[kk], false, bfrag_lds(bufRow, kk), (short)0, acc, false, false);
    const float bias = b2[net * 512 + nb + lanelo];
#pragma unroll
    for (int v = 0; v < 8; ++v) {
      float val = fmaxf(acc[v] + bias, 0.0f);
      *(unsigned short*)(r0 + (wrow + v + 8 * laneh) * H_STRIDE +
                         (nb + lanelo) * 2) = (unsigned short)f2bf(val);
    }
    wait_async0();
    __syncthreads();
  }

  // ===================== Layer 3: K=512, Nout=128 =====================
  stage_issue(Wn3, 0, 512, 6, st, tid);
  v16bf a3[16];
  {
    const char* base = r0 + (wrow + lanelo) * H_STRIDE + laneh * 16;
#pragma unroll
    for (int kk = 0; kk < 16; ++kk) {
      I4x2 p;
      p.a = *(const int4*)(base + kk * 64);
      p.b = *(const int4*)(base + kk * 64 + 32);
      a3[kk] = __builtin_bit_cast(v16bf, p);
    }
  }
  wait_async0();
  __syncthreads();
  for (int ct = 0; ct < 8; ++ct) {
    if (ct + 1 < 8)
      stage_issue(Wn3, (ct + 1) * 16, 512, 6, st + ((ct + 1) & 1) * STAGE_BUF, tid);
    const int nb = ct * 16;
    const char* bufRow =
        st + (ct & 1) * STAGE_BUF + lanelo * STAGE_STRIDE + laneh * 16;
    v8f acc = {};
#pragma unroll
    for (int kk = 0; kk < 16; ++kk)
      acc = __builtin_amdgcn_wmma_f32_16x16x32_bf16(
          false, a3[kk], false, bfrag_lds(bufRow, kk), (short)0, acc, false, false);
    const float bias = b3[net * 128 + nb + lanelo];
#pragma unroll
    for (int v = 0; v < 8; ++v) {
      float val = acc[v] + bias;
      out[(size_t)(rb + wrow + v + 8 * laneh) * 4096 + net * 128 + nb + lanelo] =
          val;
    }
    wait_async0();
    __syncthreads();
  }
}

// ------------------------------- launcher ----------------------------------
extern "C" void kernel_launch(void* const* d_in, const int* in_sizes, int n_in,
                              void* d_out, int out_size, void* d_ws,
                              size_t ws_size, hipStream_t stream) {
  const float* x  = (const float*)d_in[0];
  const float* W1 = (const float*)d_in[1];
  const float* b1 = (const float*)d_in[2];
  const float* W2 = (const float*)d_in[3];
  const float* b2 = (const float*)d_in[4];
  const float* W3 = (const float*)d_in[5];
  const float* b3 = (const float*)d_in[6];
  float* out = (float*)d_out;

  // bf16 weight staging in workspace (~28 MB)
  unsigned short* W1b = (unsigned short*)d_ws;                           // 8,388,608 B
  unsigned short* W2b = (unsigned short*)((char*)d_ws + 8388608);        // 16,777,216 B
  unsigned short* W3b = (unsigned short*)((char*)d_ws + 25165824);       //  4,194,304 B

  const int n1 = 32 * 512 * 256 / 4;
  const int n2 = 32 * 512 * 512 / 4;
  const int n3 = 32 * 128 * 512 / 4;
  cvt_f32_to_bf16<<<(n1 + 255) / 256, 256, 0, stream>>>(W1, (uint2*)W1b, n1);
  cvt_f32_to_bf16<<<(n2 + 255) / 256, 256, 0, stream>>>(W2, (uint2*)W2b, n2);
  cvt_f32_to_bf16<<<(n3 + 255) / 256, 256, 0, stream>>>(W3, (uint2*)W3b, n3);

  (void)hipFuncSetAttribute((const void*)mlp_ens_kernel,
                            hipFuncAttributeMaxDynamicSharedMemorySize,
                            LDS_BYTES);
  dim3 grid(4096 / BT, 32);   // (batch tiles, nets) = (32, 32)
  mlp_ens_kernel<<<grid, 256, LDS_BYTES, stream>>>(x, W1b, b1, W2b, b2, W3b,
                                                   b3, out);
}
